// LinearAttention_14817637171675
// MI455X (gfx1250) — matrix operand order repaired and checked by reading, not compiled, not run on hardware
//
#include <hip/hip_runtime.h>

// CDNA5 / gfx1250: wave32, WMMA 16x16x32 f16 -> f32 accumulate,
// async global->LDS staging (ASYNCcnt) for transform-free tiles.
typedef __attribute__((ext_vector_type(16))) _Float16 v16h;
typedef __attribute__((ext_vector_type(8)))  float    v8f;

#define N_  8
#define L_  8192
#define S_  8192
#define H_  8
#define D_  64
#define NH_ (N_ * H_)
#define EPS_ 1e-6f

#define SCHUNKS 64
#define CS_     (S_ / SCHUNKS)   // 128 s-values per chunk -> 4 WMMA k-steps of 32

// LDS row stride (pad 64 -> 68 floats; 68*4 = 272 B, still 16B aligned rows)
#define LP 68

// Workspace layout (floats): KV[NH_][64][64], then Ksum[NH_][64]
#define KV_OFF   0
#define KSUM_OFF (NH_ * 64 * 64)
#define WS_FLOATS (KSUM_OFF + NH_ * 64)

__device__ __forceinline__ float elu1(float x) {
    // elu(x) + 1 : x>0 ? x+1 : exp(x)
    return x > 0.0f ? x + 1.0f : __expf(x);
}

// Async global->LDS copy, 16B per lane (GVS addressing: saddr pair + u32 vaddr).
// writes LDS at byte offset lds_byte_off; tracked by ASYNCcnt.
__device__ __forceinline__ void async_copy_b128(unsigned lds_byte_off,
                                                unsigned gl_byte_off,
                                                const void* sbase) {
    asm volatile("global_load_async_to_lds_b128 %0, %1, %2"
                 :
                 : "v"(lds_byte_off), "v"(gl_byte_off), "s"(sbase)
                 : "memory");
}
__device__ __forceinline__ void wait_async0() {
    asm volatile("s_wait_asynccnt 0" ::: "memory");
}
// Low 32 bits of a generic LDS pointer == LDS byte offset (flat-LDS truncation).
__device__ __forceinline__ unsigned lds_off(const void* p) {
    return (unsigned)(unsigned long long)p;
}

__global__ void lin_attn_zero_ws(float* __restrict__ ws) {
    int i = blockIdx.x * blockDim.x + threadIdx.x;
    int stride = gridDim.x * blockDim.x;
    for (; i < WS_FLOATS; i += stride) ws[i] = 0.0f;
}

// ---------------------------------------------------------------------------
// Phase 1: KV[nh][d][v] = sum_s elu1(K[n,s,h,d]) * V[n,s,h,v]
//          Ksum[nh][d]  = sum_s elu1(K[n,s,h,d])
// Block: 256 threads (8 waves) covering the 4x4 grid of 16x16 tiles:
// wave w -> d-tile (w&3), v-tile pair (w>>2). Per 32-s step: V staged by
// async DMA to LDS, K staged through VGPRs (elu applied), then 2 WMMAs/wave.
// ---------------------------------------------------------------------------
__global__ void __launch_bounds__(256)
lin_attn_phase1(const float* __restrict__ keys,
                const float* __restrict__ values,
                float* __restrict__ ws) {
    __shared__ float ldsK[32 * LP];
    __shared__ float ldsV[32 * LP];

    const int nh   = blockIdx.y;
    const int n    = nh >> 3;          // H_ == 8
    const int h    = nh & 7;
    const int tid  = threadIdx.x;
    const int lane = tid & 31;
    const int wv   = tid >> 5;
    const int half_sel = (lane >> 4) & 1;   // 0: lanes 0-15, 1: lanes 16-31
    const int mrow = lane & 15;
    const int dt   = wv & 3;           // d-tile 0..3
    const int vp   = wv >> 2;          // v-tile pair 0..1
    const int vt0  = vp * 2, vt1 = vp * 2 + 1;

    float* KV   = ws + KV_OFF + (long)nh * 64 * 64;
    float* Ksum = ws + KSUM_OFF + nh * 64;

    const long baseKV = ((long)n * S_) * (H_ * D_) + (long)h * D_;
    const float* kbase = keys + baseKV;
    const float* vbase = values + baseKV;

    v8f acc0 = {};
    v8f acc1 = {};
    float ksum_part = 0.0f;

    for (int step = 0; step < CS_ / 32; ++step) {
        const int s0 = blockIdx.x * CS_ + step * 32;

        // Kick off async DMA of the 32x64 V tile into LDS (no transform needed).
        #pragma unroll
        for (int i = 0; i < 2; ++i) {
            int slot = tid + i * 256;          // 0..511 float4 slots
            int sl   = slot >> 4;              // s within tile, 0..31
            int c4   = (slot & 15) * 4;        // v, multiple of 4
            unsigned goff = (unsigned)(((s0 + sl) * (H_ * D_) + c4) * 4);
            async_copy_b128(lds_off(ldsV + sl * LP + c4), goff, vbase);
        }
        // Stage featurized K tile through VGPRs (overlaps with the async DMA).
        #pragma unroll
        for (int i = 0; i < 2; ++i) {
            int slot = tid + i * 256;
            int sl   = slot >> 4;
            int c4   = (slot & 15) * 4;
            float4 kq = *(const float4*)(kbase + (long)(s0 + sl) * (H_ * D_) + c4);
            int lb = sl * LP + c4;
            ldsK[lb + 0] = elu1(kq.x); ldsK[lb + 1] = elu1(kq.y);
            ldsK[lb + 2] = elu1(kq.z); ldsK[lb + 3] = elu1(kq.w);
        }
        wait_async0();
        __syncthreads();

        // A fragment: A[m][k] = Kfeat[s0+k][dt*16+m]  (16-bit A layout, ISA 7.12.2)
        // B fragment: B[k][v] = V[s0+k][vt*16+v]
        v16h a, b0, b1;
        #pragma unroll
        for (int j = 0; j < 16; ++j) {
            int ka = half_sel * 8 + j + ((j >= 8) ? 8 : 0);
            float kf = ldsK[ka * LP + dt * 16 + mrow];
            a[j] = (_Float16)kf;
            ksum_part += kf;                   // f32 running Ksum
            int kb = half_sel * 8 + j + ((j >= 8) ? 8 : 0);
            b0[j] = (_Float16)ldsV[kb * LP + vt0 * 16 + mrow];
            b1[j] = (_Float16)ldsV[kb * LP + vt1 * 16 + mrow];
        }
        acc0 = __builtin_amdgcn_wmma_f32_16x16x32_f16(false, a, false, b0,
                                                      (short)0, acc0, false, false);
        acc1 = __builtin_amdgcn_wmma_f32_16x16x32_f16(false, a, false, b1,
                                                      (short)0, acc1, false, false);
        __syncthreads();
    }

    // Accumulate partial tiles into global KV (C/D layout: M = r + 8*half_sel).
    #pragma unroll
    for (int r = 0; r < 8; ++r) {
        int d = dt * 16 + r + half_sel * 8;
        atomicAdd(&KV[d * 64 + vt0 * 16 + mrow], acc0[r]);
        atomicAdd(&KV[d * 64 + vt1 * 16 + mrow], acc1[r]);
    }
    // Ksum: lane m and lane m+16 hold complementary k-subsets of row m.
    float kfull = ksum_part + __shfl_xor(ksum_part, 16, 32);
    if (vp == 0 && lane < 16) {
        atomicAdd(&Ksum[dt * 16 + mrow], kfull);
    }
}

// ---------------------------------------------------------------------------
// Phase 2: out[n,l,h,v] = (Qfeat[l,:] . KV[:,v]) / (Qfeat[l,:] . Ksum + eps)
// Block: 128 threads (4 waves), each wave owns one 16-row M tile and all 4
// v-tiles (8 WMMAs). KV staged by async DMA; Q staged through VGPRs (elu).
// ---------------------------------------------------------------------------
__global__ void __launch_bounds__(128)
lin_attn_phase2(const float* __restrict__ queries,
                const float* __restrict__ ws,
                float* __restrict__ out) {
    __shared__ float ldsQ[4][16 * LP];
    __shared__ float ldsKV[64 * LP];
    __shared__ float ldsKs[64];

    const int nh   = blockIdx.y;
    const int n    = nh >> 3;
    const int h    = nh & 7;
    const int tid  = threadIdx.x;
    const int lane = tid & 31;
    const int wv   = tid >> 5;
    const int half_sel = (lane >> 4) & 1;
    const int mrow = lane & 15;
    const int l0   = blockIdx.x * 64 + wv * 16;

    const float* KV   = ws + KV_OFF + (long)nh * 64 * 64;
    const float* Ksum = ws + KSUM_OFF + nh * 64;

    // Async DMA: KV (64x64 f32) into LDS, 1024 float4 slots / 128 threads.
    #pragma unroll
    for (int i = 0; i < 8; ++i) {
        int slot = tid + i * 128;            // 0..1023
        int d  = slot >> 4;
        int c4 = (slot & 15) * 4;
        unsigned goff = (unsigned)((d * 64 + c4) * 4);
        async_copy_b128(lds_off(ldsKV + d * LP + c4), goff, KV);
    }
    if (tid < 64) ldsKs[tid] = Ksum[tid];
    // Stage this wave's 16x64 featurized Q tile: 256 float4 / 32 lanes.
    const long baseQ = ((long)n * L_) * (H_ * D_) + (long)h * D_;
    #pragma unroll
    for (int i = 0; i < 8; ++i) {
        int slot = lane + i * 32;            // 0..255
        int row  = slot >> 4;
        int c4   = (slot & 15) * 4;
        float4 q = *(const float4*)(queries + baseQ + (long)(l0 + row) * (H_ * D_) + c4);
        int lb = row * LP + c4;
        ldsQ[wv][lb + 0] = elu1(q.x); ldsQ[wv][lb + 1] = elu1(q.y);
        ldsQ[wv][lb + 2] = elu1(q.z); ldsQ[wv][lb + 3] = elu1(q.w);
    }
    wait_async0();
    __syncthreads();

    // A fragments (two k-steps: d 0..31 and 32..63) + f32 normalizer dot.
    v16h a0, a1;
    float qdot = 0.0f;
    #pragma unroll
    for (int j = 0; j < 16; ++j) {
        int d0 = half_sel * 8 + j + ((j >= 8) ? 8 : 0);
        float q0 = ldsQ[wv][mrow * LP + d0];
        a0[j] = (_Float16)q0;
        qdot += q0 * ldsKs[d0];
        int d1 = d0 + 32;
        float q1 = ldsQ[wv][mrow * LP + d1];
        a1[j] = (_Float16)q1;
        qdot += q1 * ldsKs[d1];
    }
    float qfull = qdot + __shfl_xor(qdot, 16, 32);   // full 64-d dot, row = mrow
    float sval  = 1.0f / (qfull + EPS_);

    v8f acc[4] = {};
    #pragma unroll
    for (int vt = 0; vt < 4; ++vt) {
        v16h bb0, bb1;
        #pragma unroll
        for (int j = 0; j < 16; ++j) {
            int kb = half_sel * 8 + j + ((j >= 8) ? 8 : 0);
            bb0[j] = (_Float16)ldsKV[kb * LP + vt * 16 + mrow];
            bb1[j] = (_Float16)ldsKV[(kb + 32) * LP + vt * 16 + mrow];
        }
        acc[vt] = __builtin_amdgcn_wmma_f32_16x16x32_f16(false, a0, false, bb0,
                                                         (short)0, acc[vt], false, false);
        acc[vt] = __builtin_amdgcn_wmma_f32_16x16x32_f16(false, a1, false, bb1,
                                                         (short)0, acc[vt], false, false);
    }

    // Per-row scales: row r+8*half_sel's scale lives in lane (r+8*half_sel).
    float scale8[8];
    #pragma unroll
    for (int r = 0; r < 8; ++r)
        scale8[r] = __shfl(sval, r + half_sel * 8, 32);

    #pragma unroll
    for (int vt = 0; vt < 4; ++vt) {
        #pragma unroll
        for (int r = 0; r < 8; ++r) {
            int row = r + half_sel * 8;
            out[baseQ + (long)(l0 + row) * (H_ * D_) + vt * 16 + mrow] =
                acc[vt][r] * scale8[r];
        }
    }
}

extern "C" void kernel_launch(void* const* d_in, const int* in_sizes, int n_in,
                              void* d_out, int out_size, void* d_ws, size_t ws_size,
                              hipStream_t stream) {
    const float* queries = (const float*)d_in[0];
    const float* keys    = (const float*)d_in[1];
    const float* values  = (const float*)d_in[2];
    float* out = (float*)d_out;
    float* ws  = (float*)d_ws;

    lin_attn_zero_ws<<<1040, 256, 0, stream>>>(ws);
    lin_attn_phase1<<<dim3(SCHUNKS, NH_), 256, 0, stream>>>(keys, values, ws);
    lin_attn_phase2<<<dim3(L_ / 64, NH_), 128, 0, stream>>>(queries, ws, out);
}
